// AttentiveBimodalCSRPool_55946243997764
// MI455X (gfx1250) — matrix operand order repaired and checked by reading, not compiled, not run on hardware
//
#include <hip/hip_runtime.h>
#include <hip/hip_bf16.h>
#include <math.h>

// ---------------------------------------------------------------------------
// CDNA5 (gfx1250) wave32 WMMA types
// ---------------------------------------------------------------------------
typedef __attribute__((ext_vector_type(16))) __bf16 v16bf;
typedef __attribute__((ext_vector_type(8)))  float  v8f;
typedef __attribute__((ext_vector_type(4)))  float  v4f;   // native vector for NT stores

#define FLT_BIG 3.402823466e+38f

static __device__ inline v8f wmma_bf16(v16bf a, v16bf b, v8f c) {
  // D(16x16 f32) = A(16x32 bf16) x B(32x16 bf16) + C
  return __builtin_amdgcn_wmma_f32_16x16x32_bf16(false, a, false, b, (short)0, c, false, false);
}

static __device__ inline v16bf pack_a(float4 c0, float4 c1, float4 c2, float4 c3) {
  v16bf a;
  a[0]=(__bf16)c0.x; a[1]=(__bf16)c0.y; a[2]=(__bf16)c0.z; a[3]=(__bf16)c0.w;
  a[4]=(__bf16)c1.x; a[5]=(__bf16)c1.y; a[6]=(__bf16)c1.z; a[7]=(__bf16)c1.w;
  a[8]=(__bf16)c2.x; a[9]=(__bf16)c2.y; a[10]=(__bf16)c2.z; a[11]=(__bf16)c2.w;
  a[12]=(__bf16)c3.x; a[13]=(__bf16)c3.y; a[14]=(__bf16)c3.z; a[15]=(__bf16)c3.w;
  return a;
}

// 16-bit A-matrix 16x32 layout (ISA 7.12.2):
//   lane L (m = L&15, koff = (L>=16)?8:0):
//   elements 0..7  <-> K = kbase + koff + 0..7
//   elements 8..15 <-> K = kbase + 16 + koff + 0..7
static __device__ inline v16bf load_a_plain(const float* __restrict__ src, int row,
                                            int ld, int kbase, int lane) {
  int koff = (lane >> 4) * 8;
  const float* p = src + (size_t)row * ld + kbase + koff;
  float4 c0 = *(const float4*)(p);
  float4 c1 = *(const float4*)(p + 4);
  float4 c2 = *(const float4*)(p + 16);
  float4 c3 = *(const float4*)(p + 20);
  return pack_a(c0, c1, c2, c3);
}

static __device__ inline float relaf(float v, float s, float h) {
  return fmaxf(fmaf(v, s, h), 0.0f);
}
static __device__ inline float4 ar4(float4 c, float4 s, float4 h) {
  return make_float4(relaf(c.x,s.x,h.x), relaf(c.y,s.y,h.y),
                     relaf(c.z,s.z,h.z), relaf(c.w,s.w,h.w));
}

// A fragment with fused BatchNorm affine + ReLU: a = relu(src*scale[k]+shift[k])
static __device__ inline v16bf load_a_affrelu(const float* __restrict__ src, int row,
                                              int ld, int kbase,
                                              const float* sc, const float* sh, int lane) {
  int koff = (lane >> 4) * 8;
  int kk = kbase + koff;
  const float* p = src + (size_t)row * ld + kk;
  float4 c0 = *(const float4*)(p);
  float4 c1 = *(const float4*)(p + 4);
  float4 c2 = *(const float4*)(p + 16);
  float4 c3 = *(const float4*)(p + 20);
  float4 s0 = *(const float4*)(sc + kk);
  float4 s1 = *(const float4*)(sc + kk + 4);
  float4 s2 = *(const float4*)(sc + kk + 16);
  float4 s3 = *(const float4*)(sc + kk + 20);
  float4 h0 = *(const float4*)(sh + kk);
  float4 h1 = *(const float4*)(sh + kk + 4);
  float4 h2 = *(const float4*)(sh + kk + 16);
  float4 h3 = *(const float4*)(sh + kk + 20);
  return pack_a(ar4(c0,s0,h0), ar4(c1,s1,h1), ar4(c2,s2,h2), ar4(c3,s3,h3));
}

// 16-bit B-matrix 32x16 layout: lane L (n = L&15), koff = (L>=16)?8:0;
// element i<8 <-> K = kbase + koff + i, i>=8 <-> K = kbase + 16 + koff + (i-8).
static __device__ inline v16bf load_b_frag(const float* __restrict__ W, int ldw,
                                           int kbase, int n0, int lane) {
  int n  = n0 + (lane & 15);
  int kb = kbase + ((lane >> 4) * 8);
  v16bf b;
#pragma unroll
  for (int i = 0; i < 16; ++i) b[i] = (__bf16)W[(size_t)(kb + i + (i >= 8 ? 8 : 0)) * ldw + n];
  return b;
}

// ---------------------------------------------------------------------------
// Kernels
// ---------------------------------------------------------------------------
__global__ void zero_f32_kernel(float* p, int n) {
  int i = blockIdx.x * blockDim.x + threadIdx.x;
  if (i < n) p[i] = 0.0f;
}

// seg[v] = s with csr[s] <= v < csr[s+1] (upper_bound - 1; empty segs skipped)
__global__ void seg_build_kernel(const int* __restrict__ csr, int* __restrict__ seg,
                                 int V, int N) {
  int v = blockIdx.x * blockDim.x + threadIdx.x;
  if (v >= V) return;
  int lo = 0, hi = N + 1;               // search over csr[0..N]
  while (lo < hi) {
    int mid = (lo + hi) >> 1;
    if (csr[mid] <= v) lo = mid + 1; else hi = mid;
  }
  seg[v] = lo - 1;
}

// out[rows,64] = A[rows,kin] @ W[kin,64] + bias, bf16 WMMA, f32 accumulate.
// Optionally accumulates per-column sum / sumsq into stats[0:64] / stats[64:128].
__global__ void gemm_bias_kernel(const float* __restrict__ A, const float* __restrict__ W,
                                 const float* __restrict__ bias, float* __restrict__ out,
                                 int rows, int kin, float* stats) {
  __shared__ float sSum[64];
  __shared__ float sSq[64];
  const int lane = threadIdx.x & 31;
  const int wave = threadIdx.x >> 5;
  const int mtile = wave >> 2, ntile = wave & 3;
  const int n0 = ntile * 16;
  const int blockRow = blockIdx.x * 32;
  const int m0 = blockRow + mtile * 16;
  if (stats) {
    if (threadIdx.x < 64) { sSum[threadIdx.x] = 0.0f; sSq[threadIdx.x] = 0.0f; }
    __syncthreads();
  }
  const int col = n0 + (lane & 15);
  const float bval = bias[col];
  v8f c;
#pragma unroll
  for (int r = 0; r < 8; ++r) c[r] = bval;

  int rowA = m0 + (lane & 15);
  if (rowA >= rows) rowA = rows - 1;
  const int ksteps = kin >> 5;
  for (int ks = 0; ks < ksteps; ++ks) {
    v16bf a = load_a_plain(A, rowA, kin, ks * 32, lane);
    v16bf b = load_b_frag(W, 64, ks * 32, n0, lane);
    c = wmma_bf16(a, b, c);
  }

  // D store: one base pointer per lane, r*64 immediate offsets.
  const int half = (lane >> 4) * 8;
  const int mBase = m0 + half;
  float* po = out + (size_t)mBase * 64 + col;
  if (m0 + 16 <= rows) {                       // full tile in range (common)
#pragma unroll
    for (int r = 0; r < 8; ++r) po[(size_t)r * 64] = c[r];
    if (stats) {
      float s = 0.0f, s2 = 0.0f;
#pragma unroll
      for (int r = 0; r < 8; ++r) { s += c[r]; s2 += c[r] * c[r]; }
      atomicAdd(&sSum[col], s);
      atomicAdd(&sSq[col], s2);
    }
  } else {                                     // tail block only
    float s = 0.0f, s2 = 0.0f;
#pragma unroll
    for (int r = 0; r < 8; ++r) {
      if (mBase + r < rows) {
        po[(size_t)r * 64] = c[r];
        s += c[r]; s2 += c[r] * c[r];
      }
    }
    if (stats) { atomicAdd(&sSum[col], s); atomicAdd(&sSq[col], s2); }
  }
  if (stats) {
    __syncthreads();
    if (threadIdx.x < 64) {
      atomicAdd(&stats[threadIdx.x], sSum[threadIdx.x]);
      atomicAdd(&stats[64 + threadIdx.x], sSq[threadIdx.x]);
    }
  }
}

// BN finalize: scale = g*rstd ; shift = beta - mean*scale
__global__ void bn_finalize_kernel(const float* __restrict__ stats,
                                   const float* __restrict__ g, const float* __restrict__ beta,
                                   float* __restrict__ scale, float* __restrict__ shift,
                                   float invV) {
  int j = threadIdx.x;              // 64 threads
  float mu = stats[j] * invV;
  float var = stats[64 + j] * invV - mu * mu;
  float rstd = rsqrtf(var + 1e-5f);
  float sc = g[j] * rstd;
  scale[j] = sc;
  shift[j] = beta[j] - mu * sc;
}

// In-place layer 2: y <- relu(BN1(y)) @ W2 + b2, with stats of the output.
// rows must be a multiple of 32 (V is).
__global__ void gemm_affine_relu_kernel(float* __restrict__ y,
                                        const float* __restrict__ scale,
                                        const float* __restrict__ shift,
                                        const float* __restrict__ W,
                                        const float* __restrict__ bias,
                                        int rows, float* stats) {
  __shared__ __align__(16) float sScale[64];
  __shared__ __align__(16) float sShift[64];
  __shared__ float sSum[64];
  __shared__ float sSq[64];
  if (threadIdx.x < 64) {
    sScale[threadIdx.x] = scale[threadIdx.x];
    sShift[threadIdx.x] = shift[threadIdx.x];
    sSum[threadIdx.x] = 0.0f;
    sSq[threadIdx.x] = 0.0f;
  }
  __syncthreads();
  const int lane = threadIdx.x & 31;
  const int wave = threadIdx.x >> 5;
  const int mtile = wave >> 2, ntile = wave & 3;
  const int n0 = ntile * 16;
  const int blockRow = blockIdx.x * 32;
  const int m0 = blockRow + mtile * 16;
  const int col = n0 + (lane & 15);
  const int rowA = m0 + (lane & 15);

  // Load ALL inputs of this block before any in-place store (other waves in
  // this block read the same y rows that we will overwrite below).
  v16bf a0 = load_a_affrelu(y, rowA, 64, 0,  sScale, sShift, lane);
  v16bf a1 = load_a_affrelu(y, rowA, 64, 32, sScale, sShift, lane);
  v16bf b0 = load_b_frag(W, 64, 0,  n0, lane);
  v16bf b1 = load_b_frag(W, 64, 32, n0, lane);
  asm volatile("s_wait_loadcnt 0" ::: "memory");   // loads done before barrier
  __syncthreads();                                  // ...before any wave stores

  const float bval = bias[col];
  v8f c;
#pragma unroll
  for (int r = 0; r < 8; ++r) c[r] = bval;
  c = wmma_bf16(a0, b0, c);
  c = wmma_bf16(a1, b1, c);

  const int half = (lane >> 4) * 8;
  float* po = y + (size_t)(m0 + half) * 64 + col;
  float s = 0.0f, s2 = 0.0f;
#pragma unroll
  for (int r = 0; r < 8; ++r) {
    po[(size_t)r * 64] = c[r];
    s += c[r]; s2 += c[r] * c[r];
  }
  atomicAdd(&sSum[col], s);
  atomicAdd(&sSq[col], s2);
  __syncthreads();
  if (threadIdx.x < 64) {
    atomicAdd(&stats[threadIdx.x], sSum[threadIdx.x]);
    atomicAdd(&stats[64 + threadIdx.x], sSq[threadIdx.x]);
  }
}

// Fused: K[v,:] = relu(BN2(y[v,:])) @ Wk[0:64,:] + x_mod[v,:] @ Wk[64:192,:] + bk
// then X[v] = K[v,:] . Q[seg[v],:]   (K never hits HBM: D tiles -> LDS -> dot)
// rows must be a multiple of 32 (V is).
__global__ void kx_fused_kernel(const float* __restrict__ y,
                                const float* __restrict__ x_mod,
                                const float* __restrict__ Wk,
                                const float* __restrict__ bk,
                                const float* __restrict__ scale,
                                const float* __restrict__ shift,
                                const float* __restrict__ Q,
                                const int* __restrict__ seg,
                                float* __restrict__ X, int rows) {
  __shared__ __align__(16) float sScale[64];
  __shared__ __align__(16) float sShift[64];
  __shared__ __align__(16) float sK[32 * 68];     // 32 rows x 64 cols, padded
  if (threadIdx.x < 64) {
    sScale[threadIdx.x] = scale[threadIdx.x];
    sShift[threadIdx.x] = shift[threadIdx.x];
  }
  __syncthreads();
  const int lane = threadIdx.x & 31;
  const int wave = threadIdx.x >> 5;
  const int mtile = wave >> 2, ntile = wave & 3;
  const int n0 = ntile * 16;
  const int blockRow = blockIdx.x * 32;
  const int m0loc = mtile * 16;
  const int rowA = blockRow + m0loc + (lane & 15);
  const int col = n0 + (lane & 15);

  const float bval = bk[col];
  v8f c;
#pragma unroll
  for (int r = 0; r < 8; ++r) c[r] = bval;
  // K-steps 0..1: hidden features (affine+relu of y)
#pragma unroll
  for (int ks = 0; ks < 2; ++ks) {
    v16bf a = load_a_affrelu(y, rowA, 64, ks * 32, sScale, sShift, lane);
    v16bf b = load_b_frag(Wk, 64, ks * 32, n0, lane);
    c = wmma_bf16(a, b, c);
  }
  // K-steps 2..5: x_mod features (k' = 0..127 maps to Wk rows 64..191)
#pragma unroll
  for (int ks = 0; ks < 4; ++ks) {
    v16bf a = load_a_plain(x_mod, rowA, 128, ks * 32, lane);
    v16bf b = load_b_frag(Wk, 64, 64 + ks * 32, n0, lane);
    c = wmma_bf16(a, b, c);
  }
  const int half = (lane >> 4) * 8;
  float* pk = &sK[(m0loc + half) * 68 + col];
#pragma unroll
  for (int r = 0; r < 8; ++r) pk[r * 68] = c[r];
  __syncthreads();

  // X phase: 8 lanes per row, each lane dots 8 consecutive columns with Q[seg]
  const int lrow = threadIdx.x >> 3;        // 0..31
  const int cb = (threadIdx.x & 7) * 8;     // 0..56
  const int gr = blockRow + lrow;
  const int s = seg[gr];
  const float* qp = Q + (size_t)s * 64 + cb;
  float4 q0 = *(const float4*)(qp);
  float4 q1 = *(const float4*)(qp + 4);
  const float* kp = &sK[lrow * 68 + cb];
  float4 k0 = *(const float4*)(kp);
  float4 k1 = *(const float4*)(kp + 4);
  float p = k0.x*q0.x + k0.y*q0.y + k0.z*q0.z + k0.w*q0.w
          + k1.x*q1.x + k1.y*q1.y + k1.z*q1.z + k1.w*q1.w;
  p += __shfl_xor(p, 1, 8);
  p += __shfl_xor(p, 2, 8);
  p += __shfl_xor(p, 4, 8);
  if ((threadIdx.x & 7) == 0) X[gr] = p;
}

// Per-segment softmax stats: segp[s] = {mx, rsqrt(cnt), 1/(sum_e+eps), tanh(relu(mx))}
__global__ void segstat_kernel(const int* __restrict__ csr, const float* __restrict__ X,
                               float4* __restrict__ segp, int N) {
  int s = blockIdx.x * blockDim.x + threadIdx.x;
  if (s >= N) return;
  int b = csr[s], e = csr[s + 1];
  int cnt = e - b;
  if (cnt <= 0) { segp[s] = make_float4(0.f, 0.f, 0.f, 0.f); return; }
  float mx = -FLT_BIG;
  for (int v = b; v < e; ++v) mx = fmaxf(mx, X[v]);
  float rsc = rsqrtf((float)cnt);
  float sum = 0.0f;
  for (int v = b; v < e; ++v) sum += expf((X[v] - mx) * rsc);
  float inv = 1.0f / (sum + 1e-12f);
  float G = tanhf(fmaxf(mx, 0.0f));
  segp[s] = make_float4(mx, rsc, inv, G);
}

// One wave per segment: x_pool[s,f] = G * max_v( x_mod[v,f] * A[v] ); seen flag.
__global__ void pool_kernel(const int* __restrict__ csr, const float* __restrict__ X,
                            const float4* __restrict__ segp,
                            const float* __restrict__ x_mod,
                            float* __restrict__ out_pool, float* __restrict__ out_seen,
                            int N) {
  const int lane = threadIdx.x & 31;
  const int s = (blockIdx.x * blockDim.x + threadIdx.x) >> 5;
  if (s >= N) return;
  const int b = csr[s], e = csr[s + 1];
  const float4 sp = segp[s];   // {mx, rsc, inv, G}
  const int f = lane * 4;
  float4 acc = make_float4(-FLT_BIG, -FLT_BIG, -FLT_BIG, -FLT_BIG);
  for (int v = b; v < e; ++v) {
    float a = expf((X[v] - sp.x) * sp.y) * sp.z;
    float4 xm = *(const float4*)(x_mod + (size_t)v * 128 + f);
    acc.x = fmaxf(acc.x, xm.x * a);
    acc.y = fmaxf(acc.y, xm.y * a);
    acc.z = fmaxf(acc.z, xm.z * a);
    acc.w = fmaxf(acc.w, xm.w * a);
  }
  v4f res;
  if (e > b) res = (v4f){acc.x * sp.w, acc.y * sp.w, acc.z * sp.w, acc.w * sp.w};
  else       res = (v4f){0.f, 0.f, 0.f, 0.f};
  // x_pool is written once and never re-read: bypass near caches (NT store).
  __builtin_nontemporal_store(res, (v4f*)(out_pool + (size_t)s * 128 + f));
  if (lane == 0) out_seen[s] = (e > b) ? 1.0f : 0.0f;
}

// ---------------------------------------------------------------------------
// Host launcher
// ---------------------------------------------------------------------------
extern "C" void kernel_launch(void* const* d_in, const int* in_sizes, int n_in,
                              void* d_out, int out_size, void* d_ws, size_t ws_size,
                              hipStream_t stream) {
  const float* x_main = (const float*)d_in[0];
  const float* x_mod  = (const float*)d_in[1];
  const float* x_proj = (const float*)d_in[2];
  const int*   csr    = (const int*)  d_in[3];
  const float* Wq  = (const float*)d_in[4];
  const float* bq  = (const float*)d_in[5];
  const float* W1  = (const float*)d_in[6];
  const float* b1  = (const float*)d_in[7];
  const float* g1  = (const float*)d_in[8];
  const float* be1 = (const float*)d_in[9];
  const float* W2  = (const float*)d_in[10];
  const float* b2  = (const float*)d_in[11];
  const float* g2  = (const float*)d_in[12];
  const float* be2 = (const float*)d_in[13];
  const float* Wk  = (const float*)d_in[14];
  const float* bk  = (const float*)d_in[15];

  const int N = in_sizes[0] / 64;     // x_main is [N,64]
  const int V = in_sizes[1] / 128;    // x_mod  is [V,128]

  char* ws = (char*)d_ws;
  size_t off = 0;
  auto alloc = [&](size_t bytes) -> void* {
    void* p = ws + off;
    off += (bytes + 255) & ~(size_t)255;
    return p;
  };
  int*    seg  = (int*)   alloc((size_t)V * sizeof(int));
  float*  y    = (float*) alloc((size_t)V * 64 * sizeof(float));
  float*  Q    = (float*) alloc((size_t)N * 64 * sizeof(float));
  float*  X    = (float*) alloc((size_t)V * sizeof(float));
  float4* segp = (float4*)alloc((size_t)N * sizeof(float4));
  float*  par  = (float*) alloc(512 * sizeof(float));
  float* stats1 = par;        float* stats2 = par + 128;
  float* scale1 = par + 256;  float* shift1 = par + 320;
  float* scale2 = par + 384;  float* shift2 = par + 448;

  float* out_pool = (float*)d_out;
  float* out_seen = (float*)d_out + (size_t)N * 128;

  zero_f32_kernel<<<1, 256, 0, stream>>>(par, 256);
  seg_build_kernel<<<(V + 255) / 256, 256, 0, stream>>>(csr, seg, V, N);
  // Q = x_main @ Wq + bq
  gemm_bias_kernel<<<(N + 31) / 32, 256, 0, stream>>>(x_main, Wq, bq, Q, N, 64, nullptr);
  // y1 = x_proj @ W1 + b1 (+ BN1 stats)
  gemm_bias_kernel<<<(V + 31) / 32, 256, 0, stream>>>(x_proj, W1, b1, y, V, 32, stats1);
  bn_finalize_kernel<<<1, 64, 0, stream>>>(stats1, g1, be1, scale1, shift1, 1.0f / (float)V);
  // y2 = relu(BN1(y1)) @ W2 + b2 (in place, + BN2 stats)
  gemm_affine_relu_kernel<<<(V + 31) / 32, 256, 0, stream>>>(y, scale1, shift1, W2, b2, V, stats2);
  bn_finalize_kernel<<<1, 64, 0, stream>>>(stats2, g2, be2, scale2, shift2, 1.0f / (float)V);
  // K = [relu(BN2(y2)), x_mod] @ Wk + bk ; X = sum(K * Q[seg], axis=1)  (fused)
  kx_fused_kernel<<<(V + 31) / 32, 256, 0, stream>>>(y, x_mod, Wk, bk, scale2, shift2,
                                                     Q, seg, X, V);
  segstat_kernel<<<(N + 255) / 256, 256, 0, stream>>>(csr, X, segp, N);
  pool_kernel<<<(N + 7) / 8, 256, 0, stream>>>(csr, X, segp, x_mod, out_pool, out_seen, N);
}
